// PreconditionedAttention_62113817035127
// MI455X (gfx1250) — matrix operand order repaired and checked by reading, not compiled, not run on hardware
//
#include <hip/hip_runtime.h>

typedef __attribute__((ext_vector_type(16))) _Float16 v16h;
typedef __attribute__((ext_vector_type(8)))  float    v8f;
typedef __attribute__((ext_vector_type(4)))  float    v4f;

constexpr int Bb = 2, Hh = 16, S = 2048, D = 64;
constexpr int BH = Bb * Hh;
constexpr int QT = 16;          // query rows per block
constexpr int NWAVES = 4;       // waves per block
constexpr int LDST = S + 4;     // padded LDS row stride (floats): 2052 % 64 == 4 -> conflict-free

__global__ __launch_bounds__(128) void attn_fused_kernel(
    const float* __restrict__ Q,
    const float* __restrict__ K,
    const float* __restrict__ V,
    float* __restrict__ outP,
    float* __restrict__ attnW)
{
    __shared__ float sm[QT * LDST];        // exp(scores - max), per-row strip
    __shared__ float rowscale[QT];         // 1 / sum(exp)
    __shared__ float rowsq[NWAVES][QT];    // per-wave partial sum of squares

    const int tid  = threadIdx.x;
    const int wave = tid >> 5;
    const int lane = tid & 31;
    const int hlf  = lane >> 4;            // which 16-lane half of the wave
    const int l16  = lane & 15;

    const int bh = blockIdx.x >> 7;        // 128 q-tiles per (b,h)
    const int q0 = (blockIdx.x & 127) << 4;

    const size_t bhbase = (size_t)bh * S * D;
    const float* Qb = Q + bhbase + (size_t)q0 * D;
    const float* Kb = K + bhbase;
    const float* Vb = V + bhbase;

    // ---- Q A-fragments (16x32 f16 layout), reused for all of pass A ----
    // A elem j (lane): row m = l16, k = hlf*16 + j  (+32 for 2nd chunk)
    v16h aq0, aq1;
    {
        const float* qrow = Qb + l16 * D;
        #pragma unroll
        for (int g = 0; g < 2; ++g) {
            const int kb = hlf * 16 + g * 8;
            #pragma unroll
            for (int jj = 0; jj < 8; ++jj) {
                aq0[g * 8 + jj] = (_Float16)qrow[kb + jj];
                aq1[g * 8 + jj] = (_Float16)qrow[32 + kb + jj];
            }
        }
    }

    // ---- Pass A: S[m][n] = (Q Kt)/8 for this 16-row strip ----
    for (int kt = wave; kt < S / 16; kt += NWAVES) {
        const int k0 = kt * 16;
        // B elem j (lane): kk = hlf*16 + j, n = l16  ->  Kmat[k0+n][f0 + kk]
        v16h b0, b1;
        const float* krow = Kb + (size_t)(k0 + l16) * D + hlf * 16;
        #pragma unroll
        for (int j = 0; j < 16; ++j) {
            b0[j] = (_Float16)krow[j];
            b1[j] = (_Float16)krow[32 + j];
        }
        v8f c = {};
        c = __builtin_amdgcn_wmma_f32_16x16x32_f16(false, aq0, false, b0, (short)0, c, false, false);
        c = __builtin_amdgcn_wmma_f32_16x16x32_f16(false, aq1, false, b1, (short)0, c, false, false);
        // D elem r (lane): row = r + hlf*8, col = l16
        #pragma unroll
        for (int r = 0; r < 8; ++r)
            sm[(r + hlf * 8) * LDST + k0 + l16] = c[r] * 0.125f;
    }
    __syncthreads();

    // ---- row softmax: 8 threads per row, 256 cols each ----
    {
        const int row  = tid >> 3;
        const int part = tid & 7;
        float* srow = sm + row * LDST + part * 256;
        float mx = -3.4e38f;
        #pragma unroll 8
        for (int c = 0; c < 256; ++c) mx = fmaxf(mx, srow[c]);
        #pragma unroll
        for (int msk = 1; msk < 8; msk <<= 1) mx = fmaxf(mx, __shfl_xor(mx, msk, 32));
        float sum = 0.f;
        #pragma unroll 4
        for (int c = 0; c < 256; ++c) { float e = __expf(srow[c] - mx); srow[c] = e; sum += e; }
        #pragma unroll
        for (int msk = 1; msk < 8; msk <<= 1) sum += __shfl_xor(sum, msk, 32);
        if (part == 0) rowscale[row] = 1.0f / sum;
    }
    __syncthreads();

    // ---- write normalized attention weights ----
    // 536 MB total, write-once / never re-read: non-temporal stores so the
    // streamed weights do not evict the L2-resident K/V working sets.
    {
        float* wbase = attnW + ((size_t)bh * S + q0) * S;
        for (int idx = tid; idx < QT * (S / 4); idx += 128) {
            const int row = idx >> 9;         // S/4 = 512 float4 per row
            const int c4  = idx & 511;
            v4f v4 = *(const v4f*)&sm[row * LDST + c4 * 4];
            v4 *= rowscale[row];
            __builtin_nontemporal_store(v4, (v4f*)&wbase[(size_t)row * S + c4 * 4]);
        }
    }

    // ---- Pass B: out = P @ V; wave owns output cols [n0, n0+16) ----
    const int n0 = wave * 16;
    const float rs = rowscale[l16];            // A rows: m = l16
    v8f acc = {};
    for (int kc = 0; kc < S / 32; ++kc) {
        v16h a;
        const float* prow = sm + l16 * LDST + kc * 32 + hlf * 16;
        #pragma unroll
        for (int g = 0; g < 2; ++g)
            #pragma unroll
            for (int jj = 0; jj < 8; ++jj)
                a[g * 8 + jj] = (_Float16)(prow[g * 8 + jj] * rs);
        v16h b;
        const float* vcol = Vb + (size_t)(kc * 32 + hlf * 16) * D + n0 + l16;
        #pragma unroll
        for (int j = 0; j < 16; ++j)
            b[j] = (_Float16)vcol[(size_t)j * D];
        acc = __builtin_amdgcn_wmma_f32_16x16x32_f16(false, a, false, b, (short)0, acc, false, false);
    }

    // ---- row sum-of-squares: reduce 16 cols in-wave, then across waves ----
    float sq[8];
    #pragma unroll
    for (int r = 0; r < 8; ++r) sq[r] = acc[r] * acc[r];
    #pragma unroll
    for (int msk = 1; msk < 16; msk <<= 1)
        #pragma unroll
        for (int r = 0; r < 8; ++r) sq[r] += __shfl_xor(sq[r], msk, 32);
    if (l16 == 0)
        #pragma unroll
        for (int r = 0; r < 8; ++r) rowsq[wave][hlf * 8 + r] = sq[r];
    __syncthreads();

    #pragma unroll
    for (int r = 0; r < 8; ++r) {
        const int row = hlf * 8 + r;
        const float tot = rowsq[0][row] + rowsq[1][row] + rowsq[2][row] + rowsq[3][row];
        const float nrm = sqrtf(tot);
        const float val = acc[r] / (nrm + 1e-8f);
        __builtin_nontemporal_store(val, &outP[((size_t)bh * S + q0 + row) * D + n0 + l16]);
    }
}

extern "C" void kernel_launch(void* const* d_in, const int* in_sizes, int n_in,
                              void* d_out, int out_size, void* d_ws, size_t ws_size,
                              hipStream_t stream) {
    (void)in_sizes; (void)n_in; (void)out_size; (void)d_ws; (void)ws_size;
    const float* q = (const float*)d_in[0];
    const float* k = (const float*)d_in[1];
    const float* v = (const float*)d_in[2];
    float* out  = (float*)d_out;
    float* attn = out + (size_t)BH * S * D;   // tuple: (precond_out, attn_weights) flat
    dim3 grid(BH * (S / 16));                 // 4096 blocks, 1 query tile each
    attn_fused_kernel<<<grid, 128, 0, stream>>>(q, k, v, out, attn);
}
